// FlashAttention_37366215475914
// MI455X (gfx1250) — matrix operand order repaired and checked by reading, not compiled, not run on hardware
//
#include <hip/hip_runtime.h>
#include <hip/hip_bf16.h>
#include <stdint.h>

typedef __attribute__((ext_vector_type(16))) __bf16 v16bf;
typedef __attribute__((ext_vector_type(8)))  float  v8f;

constexpr int kB = 2, kT = 2048, kS = 2048, kH = 16, kD = 64, kE = 1024;
constexpr int kBT = kB * kT;            // 4096
constexpr int kBS = kB * kS;            // 4096
// softmax done in base-2 domain: score * (1/sqrt(D)) * log2(e)
constexpr float kScaleLog2 = 0.125f * 1.44269504088896340736f;

__device__ __forceinline__ unsigned short f2bf(float x) {
  union { float f; unsigned int u; } c; c.f = x;
  unsigned int r = c.u + 0x7FFFu + ((c.u >> 16) & 1u);  // round-to-nearest-even
  return (unsigned short)(r >> 16);
}

struct FragBF {
  union { v16bf v; uint4 q[2]; unsigned short h[16]; };
};

// A-matrix fragment (16x32, MxK), bf16, row-major source with leading dim ld.
// lane L: row = m0 + (L&15); elems 0..7 -> K = k0 + (L>>4)*8 + i
//                            elems 8..15 -> K = k0 + 16 + (L>>4)*8 + i
__device__ __forceinline__ FragBF load_a_frag(const unsigned short* base, int ld,
                                              int m0, int k0, int lane) {
  FragBF f;
  const int row  = m0 + (lane & 15);
  const int half = lane >> 4;
  const unsigned short* p = base + (long)row * ld + k0 + half * 8;
  f.q[0] = *(const uint4*)(p);
  f.q[1] = *(const uint4*)(p + 16);
  return f;
}

// B-matrix fragment (32x16, KxN) where B[k][n] = Wrow[n][k], Wrow row-major (N,K), ld = K-stride.
// lane L: n = n0 + (L&15); elems 0..7 -> K = k0 + (L>>4)*8 + i, elems 8..15 -> K = k0 + 16 + (L>>4)*8 + i
__device__ __forceinline__ FragBF load_b_frag(const unsigned short* Wrow, long ld,
                                              int n0, int k0, int lane) {
  FragBF f;
  const int n    = n0 + (lane & 15);
  const int half = lane >> 4;
  const unsigned short* p = Wrow + (long)n * ld + k0 + half * 8;
  f.q[0] = *(const uint4*)(p);
  f.q[1] = *(const uint4*)(p + 16);
  return f;
}

__device__ __forceinline__ v8f wmma_bf16(const FragBF& a, const FragBF& b, v8f c) {
  return __builtin_amdgcn_wmma_f32_16x16x32_bf16(false, a.v, false, b.v,
                                                 (short)0, c, false, false);
}

// ---------------------------------------------------------------- converts
__global__ __launch_bounds__(256) void cvt_f32_bf16(const float* __restrict__ src,
                                                    unsigned short* __restrict__ dst, long n) {
  long i = (long)blockIdx.x * blockDim.x + threadIdx.x;
  if (i < n) dst[i] = f2bf(src[i]);
}

// kv (B,S,2,H,D) -> kbf (B*S,E), vbf (B*S,E)
__global__ __launch_bounds__(256) void split_kv(const float* __restrict__ kv,
                                                unsigned short* __restrict__ kbf,
                                                unsigned short* __restrict__ vbf, long n) {
  long i = (long)blockIdx.x * blockDim.x + threadIdx.x;
  if (i >= n) return;
  long e = i % kE, bs = i / kE;
  kbf[i] = f2bf(kv[(bs * 2 + 0) * kE + e]);
  vbf[i] = f2bf(kv[(bs * 2 + 1) * kE + e]);
}

// ---------------------------------------------------------------- GEMM: out = A(M,K) @ W(N,K)^T + bias
// Software-pipelined: fragments for k-step i+1 are issued before the WMMAs of
// k-step i, so load latency overlaps the matrix pipe instead of s_wait 0 stalls.
// mode 0: write bf16 permuted to (B,H,Rows,D)   (Q/K projections)
// mode 1: write bf16 permuted to (B,H,D,Rows)   (V projection, transposed)
// mode 2: write f32 row-major (M,N) to fout     (output projection)
__global__ __launch_bounds__(256) void gemm_bf16(const unsigned short* __restrict__ A,
                                                 const unsigned short* __restrict__ W,
                                                 const float* __restrict__ bias,
                                                 unsigned short* __restrict__ obf,
                                                 float* __restrict__ fout,
                                                 int M, int N, int K, int mode, int Rows) {
  const int lane = threadIdx.x & 31;
  const int gw = blockIdx.x * (blockDim.x >> 5) + (threadIdx.x >> 5);
  const int nt64 = N >> 6;
  if (gw >= (M >> 4) * nt64) return;
  const int m0 = (gw / nt64) << 4;
  const int n0 = (gw % nt64) << 6;

  v8f c[4] = {};
  // prologue: k-step 0 fragments
  FragBF a0 = load_a_frag(A, K, m0, 0, lane);
  FragBF b0[4];
#pragma unroll
  for (int j = 0; j < 4; ++j) b0[j] = load_b_frag(W, K, n0 + 16 * j, 0, lane);

  for (int k0 = 32; k0 <= K - 32; k0 += 32) {
    // issue next k-step's loads first (one clause), then compute current
    FragBF a1 = load_a_frag(A, K, m0, k0, lane);
    FragBF b1[4];
#pragma unroll
    for (int j = 0; j < 4; ++j) b1[j] = load_b_frag(W, K, n0 + 16 * j, k0, lane);
    if (k0 + 32 < K)  // pull the k-step after next toward L2 (global_prefetch_b8)
      __builtin_prefetch(A + (long)(m0 + (lane & 15)) * K + k0 + 32, 0, 1);
#pragma unroll
    for (int j = 0; j < 4; ++j) c[j] = wmma_bf16(a0, b0[j], c[j]);
    a0 = a1;
#pragma unroll
    for (int j = 0; j < 4; ++j) b0[j] = b1[j];
  }
  // epilogue: last k-step
#pragma unroll
  for (int j = 0; j < 4; ++j) c[j] = wmma_bf16(a0, b0[j], c[j]);

  const int rb = (lane >> 4) * 8;
  const int col = lane & 15;
#pragma unroll
  for (int j = 0; j < 4; ++j) {
    const float bb = bias ? bias[n0 + 16 * j + col] : 0.0f;
#pragma unroll
    for (int r = 0; r < 8; ++r) {
      const float val = c[j][r] + bb;
      const int m = m0 + r + rb;
      const int n = n0 + 16 * j + col;
      if (mode == 2) {
        fout[(long)m * N + n] = val;
      } else {
        const int b = m / Rows, t = m % Rows, h = n >> 6, d = n & 63;
        if (mode == 0)
          obf[((long)(b * kH + h) * Rows + t) * kD + d] = f2bf(val);
        else
          obf[((long)(b * kH + h) * kD + d) * Rows + t] = f2bf(val);
      }
    }
  }
}

// ---------------------------------------------------------------- flash attention core
// one wave per (b,h, 16-row t-strip); online softmax over S in 32-col steps.
// Qp (B,H,T,D), Kp (B,H,S,D), VpT (B,H,D,S), all bf16. Writes ctxbf (B*T,E) bf16 + m/l stats.
// V fragments are loaded together with the K fragments so the softmax VALU work
// (shuffle butterflies + v_exp_f32) and the LDS transpose hide their latency.
__global__ __launch_bounds__(256) void flash_attn(const unsigned short* __restrict__ Qp,
                                                  const unsigned short* __restrict__ Kp,
                                                  const unsigned short* __restrict__ VpT,
                                                  unsigned short* __restrict__ ctxbf,
                                                  float* __restrict__ m_ws,
                                                  float* __restrict__ l_ws) {
  __shared__ __align__(16) unsigned short lds_p[8][16 * 32];
  const int lane = threadIdx.x & 31;
  const int widx = threadIdx.x >> 5;
  const int gw = blockIdx.x * 8 + widx;
  const int tstrips = kT >> 4;                 // 128
  if (gw >= kB * kH * tstrips) return;
  const int tt = gw % tstrips;
  const int bh = gw / tstrips;
  const int b = bh / kH, h = bh % kH;
  const int t0 = tt << 4;

  const unsigned short* Qb = Qp  + (long)bh * kT * kD;
  const unsigned short* Kb = Kp  + (long)bh * kS * kD;
  const unsigned short* Vb = VpT + (long)bh * kD * kS;

  const FragBF qa0 = load_a_frag(Qb, kD, t0, 0,  lane);
  const FragBF qa1 = load_a_frag(Qb, kD, t0, 32, lane);

  v8f ctx[4] = {};
  float m_r[8], l_r[8];
#pragma unroll
  for (int r = 0; r < 8; ++r) { m_r[r] = -1e30f; l_r[r] = 0.0f; }

  const int rb = (lane >> 4) * 8;
  const int col = lane & 15;

  for (int s0 = 0; s0 < kS; s0 += 32) {
    // one big load clause: 4 K-fragments + 4 V-fragments (16 x b128)
    FragBF k0a = load_b_frag(Kb, kD, s0,      0,  lane);
    FragBF k0b = load_b_frag(Kb, kD, s0,      32, lane);
    FragBF k1a = load_b_frag(Kb, kD, s0 + 16, 0,  lane);
    FragBF k1b = load_b_frag(Kb, kD, s0 + 16, 32, lane);
    FragBF vb[4];
#pragma unroll
    for (int j = 0; j < 4; ++j) vb[j] = load_b_frag(Vb, kS, 16 * j, s0, lane);

    v8f sc0 = {}, sc1 = {};
    sc0 = wmma_bf16(qa0, k0a, sc0); sc0 = wmma_bf16(qa1, k0b, sc0);
    sc1 = wmma_bf16(qa0, k1a, sc1); sc1 = wmma_bf16(qa1, k1b, sc1);

#pragma unroll
    for (int r = 0; r < 8; ++r) { sc0[r] *= kScaleLog2; sc1[r] *= kScaleLog2; }

    // online softmax update (rows r+rb live across the 16-lane column group)
#pragma unroll
    for (int r = 0; r < 8; ++r) {
      float tmax = fmaxf(sc0[r], sc1[r]);
#pragma unroll
      for (int off = 1; off <= 8; off <<= 1) tmax = fmaxf(tmax, __shfl_xor(tmax, off, 32));
      const float mn = fmaxf(m_r[r], tmax);
      const float corr = exp2f(m_r[r] - mn);
      sc0[r] = exp2f(sc0[r] - mn);
      sc1[r] = exp2f(sc1[r] - mn);
      float tsum = sc0[r] + sc1[r];
#pragma unroll
      for (int off = 1; off <= 8; off <<= 1) tsum += __shfl_xor(tsum, off, 32);
      l_r[r] = l_r[r] * corr + tsum;
      m_r[r] = mn;
#pragma unroll
      for (int j = 0; j < 4; ++j) ctx[j][r] *= corr;
    }

    // D-layout -> A-layout transpose of P through per-wave LDS tile (16x32 bf16)
#pragma unroll
    for (int r = 0; r < 8; ++r) {
      lds_p[widx][(r + rb) * 32 + col]      = f2bf(sc0[r]);
      lds_p[widx][(r + rb) * 32 + 16 + col] = f2bf(sc1[r]);
    }
    FragBF pa;
    pa.q[0] = *(const uint4*)&lds_p[widx][(lane & 15) * 32 + (lane >> 4) * 8];
    pa.q[1] = *(const uint4*)&lds_p[widx][(lane & 15) * 32 + 16 + (lane >> 4) * 8];

#pragma unroll
    for (int j = 0; j < 4; ++j)            // B[k=s][n=d] = VpT[d][s]
      ctx[j] = wmma_bf16(pa, vb[j], ctx[j]);
  }

  // normalize + store ctx as bf16 (B*T, E) for the out-projection GEMM
#pragma unroll
  for (int r = 0; r < 8; ++r) {
    const float inv = 1.0f / l_r[r];
    const int t = t0 + r + rb;
#pragma unroll
    for (int j = 0; j < 4; ++j)
      ctxbf[(long)(b * kT + t) * kE + h * kD + 16 * j + col] = f2bf(ctx[j][r] * inv);
  }
  if (col == 0) {
#pragma unroll
    for (int r = 0; r < 8; ++r) {
      const int t = t0 + r + rb;
      m_ws[(long)bh * kT + t] = m_r[r];
      l_ws[(long)bh * kT + t] = l_r[r];
    }
  }
}

// ---------------------------------------------------------------- attn weights mean over heads
// one wave per (b, 16-row t-tile, 16-col s-tile): recompute scores per head,
// normalize with stored (m,l), average over heads -> attn (B,T,S) f32.
__global__ __launch_bounds__(256) void attn_mean(const unsigned short* __restrict__ Qp,
                                                 const unsigned short* __restrict__ Kp,
                                                 const float* __restrict__ m_ws,
                                                 const float* __restrict__ l_ws,
                                                 float* __restrict__ attn) {
  const int lane = threadIdx.x & 31;
  const int gw = blockIdx.x * 8 + (threadIdx.x >> 5);
  const int sstrips = kS >> 4, tstrips = kT >> 4;
  if (gw >= kB * tstrips * sstrips) return;
  const int st = gw % sstrips;
  const int tt = (gw / sstrips) % tstrips;
  const int b = gw / (sstrips * tstrips);
  const int t0 = tt << 4, s0 = st << 4;
  const int rb = (lane >> 4) * 8;
  const int col = lane & 15;

  v8f acc = {};
  for (int h = 0; h < kH; ++h) {
    const int bh = b * kH + h;
    const unsigned short* Qb = Qp + (long)bh * kT * kD;
    const unsigned short* Kb = Kp + (long)bh * kS * kD;
    FragBF qa0 = load_a_frag(Qb, kD, t0, 0,  lane);
    FragBF qa1 = load_a_frag(Qb, kD, t0, 32, lane);
    FragBF kba = load_b_frag(Kb, kD, s0, 0,  lane);
    FragBF kbb = load_b_frag(Kb, kD, s0, 32, lane);
    v8f sc = {};
    sc = wmma_bf16(qa0, kba, sc);
    sc = wmma_bf16(qa1, kbb, sc);
#pragma unroll
    for (int r = 0; r < 8; ++r) {
      const int t = t0 + r + rb;
      const float mm = m_ws[(long)bh * kT + t];
      const float ll = l_ws[(long)bh * kT + t];
      acc[r] += exp2f(sc[r] * kScaleLog2 - mm) / ll;
    }
  }
  const float invH = 1.0f / (float)kH;
#pragma unroll
  for (int r = 0; r < 8; ++r) {
    const int t = t0 + r + rb;
    attn[(long)(b * kT + t) * kS + s0 + col] = acc[r] * invH;
  }
}

// ---------------------------------------------------------------- host
extern "C" void kernel_launch(void* const* d_in, const int* in_sizes, int n_in,
                              void* d_out, int out_size, void* d_ws, size_t ws_size,
                              hipStream_t stream) {
  const float* q          = (const float*)d_in[0];
  const float* kv         = (const float*)d_in[1];
  const float* in_proj_w  = (const float*)d_in[2];
  const float* in_proj_b  = (const float*)d_in[3];
  const float* out_proj_w = (const float*)d_in[4];
  const float* out_proj_b = (const float*)d_in[5];

  char* ws = (char*)d_ws;
  size_t off = 0;
  auto take = [&](size_t bytes) { size_t o = off; off = (off + bytes + 255) & ~(size_t)255; return o; };
  unsigned short* qbf   = (unsigned short*)(ws + take((size_t)kBT * kE * 2));
  unsigned short* kbf   = (unsigned short*)(ws + take((size_t)kBS * kE * 2));
  unsigned short* vbf   = (unsigned short*)(ws + take((size_t)kBS * kE * 2));
  unsigned short* wbf   = (unsigned short*)(ws + take((size_t)3 * kE * kE * 2));
  unsigned short* owbf  = (unsigned short*)(ws + take((size_t)kE * kE * 2));
  unsigned short* Qp    = (unsigned short*)(ws + take((size_t)kBT * kE * 2));
  unsigned short* Kp    = (unsigned short*)(ws + take((size_t)kBS * kE * 2));
  unsigned short* VpT   = (unsigned short*)(ws + take((size_t)kBS * kE * 2));
  unsigned short* ctxbf = (unsigned short*)(ws + take((size_t)kBT * kE * 2));
  float*          m_ws  = (float*)(ws + take((size_t)kB * kH * kT * 4));
  float*          l_ws  = (float*)(ws + take((size_t)kB * kH * kT * 4));

  const long nQ = (long)kBT * kE, nKV = (long)kBS * kE;
  cvt_f32_bf16<<<(nQ + 255) / 256, 256, 0, stream>>>(q, qbf, nQ);
  split_kv   <<<(nKV + 255) / 256, 256, 0, stream>>>(kv, kbf, vbf, nKV);
  cvt_f32_bf16<<<((long)3 * kE * kE + 255) / 256, 256, 0, stream>>>(in_proj_w, wbf, (long)3 * kE * kE);
  cvt_f32_bf16<<<((long)kE * kE + 255) / 256, 256, 0, stream>>>(out_proj_w, owbf, (long)kE * kE);

  // projections: waves = (M/16)*(N/64) = 4096 -> 512 blocks of 8 waves
  gemm_bf16<<<512, 256, 0, stream>>>(qbf, wbf,                       in_proj_b,          Qp,  nullptr, kBT, kE, kE, 0, kT);
  gemm_bf16<<<512, 256, 0, stream>>>(kbf, wbf + (size_t)kE * kE,     in_proj_b + kE,     Kp,  nullptr, kBS, kE, kE, 0, kS);
  gemm_bf16<<<512, 256, 0, stream>>>(vbf, wbf + (size_t)2 * kE * kE, in_proj_b + 2 * kE, VpT, nullptr, kBS, kE, kE, 1, kS);

  // flash attention: B*H*(T/16) = 4096 waves -> 512 blocks
  flash_attn<<<512, 256, 0, stream>>>(Qp, Kp, VpT, ctxbf, m_ws, l_ws);

  // out projection: f32 straight into d_out (B,T,H,D) == (B*T, E) row-major
  gemm_bf16<<<512, 256, 0, stream>>>(ctxbf, owbf, out_proj_b, nullptr, (float*)d_out, kBT, kE, kE, 2, 0);

  // averaged attention weights: B*(T/16)*(S/16) = 32768 waves -> 4096 blocks
  attn_mean<<<4096, 256, 0, stream>>>(Qp, Kp, m_ws, l_ws, (float*)d_out + (size_t)kBT * kE);
}
